// HGSurv_56908316672470
// MI455X (gfx1250) — matrix operand order repaired and checked by reading, not compiled, not run on hardware
//
#include <hip/hip_runtime.h>
#include <math.h>

// ---- problem constants --------------------------------------------------
constexpr int Bb   = 2;
constexpr int Nn   = 2048;
constexpr int IND  = 1024;
constexpr int HID  = 512;
constexpr int HEADS = 8;
constexpr int HD   = 64;
constexpr int KNN  = 9;          // K+1 neighbors (incl self)
constexpr int MT   = Bb * Nn;    // 4096 flattened rows
constexpr float EPS   = 1e-5f;
constexpr float SCALE = 0.125f;  // 1/sqrt(64)

typedef __attribute__((ext_vector_type(16))) _Float16 v16h;
typedef __attribute__((ext_vector_type(8)))  float    v8f;

__device__ __forceinline__ v8f wmma16(v16h a, v16h b, v8f c) {
  return __builtin_amdgcn_wmma_f32_16x16x32_f16(false, a, false, b, (short)0, c,
                                                false, false);
}

__device__ __forceinline__ float blk_sum256(float v, float* red) {
  int t = threadIdx.x;
  red[t] = v; __syncthreads();
  for (int s = 128; s > 0; s >>= 1) { if (t < s) red[t] += red[t + s]; __syncthreads(); }
  v = red[0]; __syncthreads();
  return v;
}
__device__ __forceinline__ float blk_max256(float v, float* red) {
  int t = threadIdx.x;
  red[t] = v; __syncthreads();
  for (int s = 128; s > 0; s >>= 1) { if (t < s) red[t] = fmaxf(red[t], red[t + s]); __syncthreads(); }
  v = red[0]; __syncthreads();
  return v;
}

// =========================================================================
// Tiled WMMA GEMM:  C[M,N] = act(A[M,K] @ W[N,K]^T + bias[N])
// 128 threads (4 waves); 64x64 output tile; K stepped by 32; f32->f16 via LDS
// ACT: 0 = identity, 1 = exact GELU, 2 = tanh
// =========================================================================
template<int ACT>
__global__ __launch_bounds__(128) void gemm_xwT(
    const float* __restrict__ A, const float* __restrict__ W,
    const float* __restrict__ bias, float* __restrict__ C,
    int M, int N, int K) {
  __shared__ _Float16 As[64][40];   // 64 rows x 32 k (+pad)
  __shared__ _Float16 Bs[32][72];   // 32 k x 64 cols (+pad)
  const int m0 = blockIdx.y * 64, n0 = blockIdx.x * 64;
  const int tid = threadIdx.x, lane = tid & 31, wave = tid >> 5;
  const int row = lane & 15, half = lane >> 4;
  v8f acc[4];
  #pragma unroll
  for (int j = 0; j < 4; ++j)
    #pragma unroll
    for (int r = 0; r < 8; ++r) acc[j][r] = 0.f;

  for (int k0 = 0; k0 < K; k0 += 32) {
    for (int t = tid; t < 64 * 32; t += 128) {          // stage A tile
      int r = t >> 5, c = t & 31;
      As[r][c] = (_Float16)A[(size_t)(m0 + r) * K + k0 + c];
    }
    for (int t = tid; t < 64 * 32; t += 128) {          // stage W tile (transposed)
      int j = t >> 5, c = t & 31;
      Bs[c][j] = (_Float16)W[(size_t)(n0 + j) * K + k0 + c];
    }
    __syncthreads();
    // A fragment: 16x32 f16, ISA layout (lanes 0-15: K 0-7/16-23; 16-31: K 8-15/24-31)
    v16h af;
    #pragma unroll
    for (int g = 0; g < 8; ++g)
      #pragma unroll
      for (int p = 0; p < 2; ++p) {
        int kk = ((g < 4) ? 0 : 16) + (half ? 8 : 0) + 2 * (g & 3) + p;
        af[g * 2 + p] = As[wave * 16 + row][kk];
      }
    #pragma unroll
    for (int jt = 0; jt < 4; ++jt) {
      v16h bf;   // B fragment 32x16: col = lane%16, K = (h&7) + 8*(lane/16) + 16*(h/8)
      #pragma unroll
      for (int h = 0; h < 16; ++h) bf[h] = Bs[(h & 7) + 8 * half + 16 * (h >> 3)][jt * 16 + row];
      acc[jt] = wmma16(af, bf, acc[jt]);
    }
    __syncthreads();
  }
  #pragma unroll
  for (int jt = 0; jt < 4; ++jt)
    #pragma unroll
    for (int r = 0; r < 8; ++r) {
      int m = m0 + wave * 16 + r + 8 * half;
      int n = n0 + jt * 16 + row;
      float v = acc[jt][r] + bias[n];
      if (ACT == 1) v = 0.5f * v * (1.f + erff(v * 0.70710678118f)); // exact gelu
      if (ACT == 2) v = tanhf(v);
      C[(size_t)m * N + n] = v;
    }
}

// =========================================================================
// Flash attention, one wave per (16-query tile, head, batch).
// QK^T and PV on WMMA; online softmax with shfl_xor row reductions.
// =========================================================================
__global__ __launch_bounds__(32) void flash_k(const float* __restrict__ qkv,
                                              float* __restrict__ ao) {
  __shared__ _Float16 Kl[32][68];
  __shared__ _Float16 Vl[32][68];
  __shared__ _Float16 Pl[16][36];
  const int q0 = blockIdx.x * 16, hh = blockIdx.y, b = blockIdx.z;
  const int lane = threadIdx.x, row = lane & 15, half = lane >> 4;
  const int RS = 3 * HID;
  const float* qb = qkv + ((size_t)b * Nn) * RS + hh * HD;
  const float* kb = qb + HID;
  const float* vb = qb + 2 * HID;

  v16h aq[2];                                  // Q tile as two 16x32 A fragments
  #pragma unroll
  for (int ks = 0; ks < 2; ++ks)
    #pragma unroll
    for (int g = 0; g < 8; ++g)
      #pragma unroll
      for (int p = 0; p < 2; ++p) {
        int kk = ((g < 4) ? 0 : 16) + (half ? 8 : 0) + 2 * (g & 3) + p + 32 * ks;
        aq[ks][g * 2 + p] = (_Float16)qb[(size_t)(q0 + row) * RS + kk];
      }
  float mrow[8], lrow[8];
  v8f O[4];
  #pragma unroll
  for (int r = 0; r < 8; ++r) { mrow[r] = -1e30f; lrow[r] = 0.f; }
  #pragma unroll
  for (int jt = 0; jt < 4; ++jt)
    #pragma unroll
    for (int r = 0; r < 8; ++r) O[jt][r] = 0.f;

  for (int kc = 0; kc < Nn; kc += 32) {
    const float* krow = kb + (size_t)(kc + lane) * RS;   // lane = key index
    const float* vrow = vb + (size_t)(kc + lane) * RS;
    #pragma unroll
    for (int d = 0; d < 64; ++d) {
      Kl[lane][d] = (_Float16)krow[d];
      Vl[lane][d] = (_Float16)vrow[d];
    }
    __syncthreads();
    v8f S[2];
    #pragma unroll
    for (int s = 0; s < 2; ++s) {
      #pragma unroll
      for (int r = 0; r < 8; ++r) S[s][r] = 0.f;
      #pragma unroll
      for (int ks = 0; ks < 2; ++ks) {
        v16h bf;                               // K^T as B fragment (d x key)
        #pragma unroll
        for (int h = 0; h < 16; ++h) bf[h] = Kl[s * 16 + row][(h & 7) + 8 * half + 16 * (h >> 3) + 32 * ks];
        S[s] = wmma16(aq[ks], bf, S[s]);
      }
    }
    #pragma unroll
    for (int r = 0; r < 8; ++r) {
      float s0 = S[0][r] * SCALE, s1 = S[1][r] * SCALE;
      float mx = fmaxf(s0, s1);
      #pragma unroll
      for (int off = 8; off >= 1; off >>= 1) mx = fmaxf(mx, __shfl_xor(mx, off, 32));
      float mnew = fmaxf(mrow[r], mx);
      float fsc  = expf(mrow[r] - mnew);
      mrow[r] = mnew;
      float p0 = expf(s0 - mnew), p1 = expf(s1 - mnew);
      float ps = p0 + p1;
      #pragma unroll
      for (int off = 8; off >= 1; off >>= 1) ps += __shfl_xor(ps, off, 32);
      lrow[r] = lrow[r] * fsc + ps;
      #pragma unroll
      for (int jt = 0; jt < 4; ++jt) O[jt][r] *= fsc;
      Pl[r + 8 * half][row]      = (_Float16)p0;
      Pl[r + 8 * half][16 + row] = (_Float16)p1;
    }
    __syncthreads();
    v16h ap;                                   // P (16x32) as A fragment
    #pragma unroll
    for (int g = 0; g < 8; ++g)
      #pragma unroll
      for (int p = 0; p < 2; ++p) {
        int kk = ((g < 4) ? 0 : 16) + (half ? 8 : 0) + 2 * (g & 3) + p;
        ap[g * 2 + p] = Pl[row][kk];
      }
    #pragma unroll
    for (int jt = 0; jt < 4; ++jt) {
      v16h bv;                                 // V as B fragment (key x d)
      #pragma unroll
      for (int h = 0; h < 16; ++h) bv[h] = Vl[(h & 7) + 8 * half + 16 * (h >> 3)][jt * 16 + row];
      O[jt] = wmma16(ap, bv, O[jt]);
    }
    __syncthreads();
  }
  #pragma unroll
  for (int jt = 0; jt < 4; ++jt)
    #pragma unroll
    for (int r = 0; r < 8; ++r) {
      int q = q0 + r + 8 * half;
      ao[((size_t)b * Nn + q) * HID + hh * HD + jt * 16 + row] = O[jt][r] / lrow[r];
    }
}

// ---- GLU: h = val * sigmoid(gate) --------------------------------------
__global__ void glu_k(const float* __restrict__ vg, float* __restrict__ h) {
  int i = blockIdx.x * 256 + threadIdx.x;   // MT*HID
  int r = i >> 9, c = i & 511;
  float val  = vg[(size_t)r * 1024 + c];
  float gate = vg[(size_t)r * 1024 + 512 + c];
  h[i] = val * (1.f / (1.f + expf(-gate)));
}

// ---- row-wise sum of squares -------------------------------------------
__global__ __launch_bounds__(256) void rowsq_k(const float* __restrict__ h,
                                               float* __restrict__ sq) {
  __shared__ float red[256];
  int r = blockIdx.x;
  float p = 0.f;
  for (int c = threadIdx.x; c < HID; c += 256) {
    float v = h[(size_t)r * HID + c]; p += v * v;
  }
  float s = blk_sum256(p, red);
  if (threadIdx.x == 0) sq[r] = s;
}

// ---- streaming top-9 nearest neighbors (no NxN matrix) ------------------
__global__ __launch_bounds__(256) void knn_k(const float* __restrict__ h,
                                             const float* __restrict__ sq,
                                             int* __restrict__ idx) {
  __shared__ float hn[HID];
  __shared__ float cd[256 * KNN];
  __shared__ int   ci[256 * KNN];
  __shared__ float rv[256];
  __shared__ int   rp[256];
  const int bn = blockIdx.x, b = bn / Nn, n = bn % Nn;
  const int tid = threadIdx.x;
  const float* hb = h + (size_t)b * Nn * HID;
  for (int c = tid; c < HID; c += 256) hn[c] = hb[(size_t)n * HID + c];
  __syncthreads();
  float bd[KNN]; int bi[KNN];
  #pragma unroll
  for (int j = 0; j < KNN; ++j) { bd[j] = 1e30f; bi[j] = 0; }
  const float sn = sq[bn];
  for (int m = tid; m < Nn; m += 256) {
    const float* hm = hb + (size_t)m * HID;
    float dot = 0.f;
    for (int c = 0; c < HID; ++c) dot += hn[c] * hm[c];
    float d2 = sn + sq[b * Nn + m] - 2.f * dot;
    float d = sqrtf(fmaxf(d2, 0.f));
    if (d < bd[KNN - 1]) {
      bd[KNN - 1] = d; bi[KNN - 1] = m;
      #pragma unroll
      for (int j = KNN - 1; j > 0; --j)
        if (bd[j] < bd[j - 1]) {
          float td = bd[j]; bd[j] = bd[j - 1]; bd[j - 1] = td;
          int   ti = bi[j]; bi[j] = bi[j - 1]; bi[j - 1] = ti;
        }
    }
  }
  for (int j = 0; j < KNN; ++j) { cd[tid * KNN + j] = bd[j]; ci[tid * KNN + j] = bi[j]; }
  __syncthreads();
  for (int sel = 0; sel < KNN; ++sel) {       // 9x block argmin
    float best = 1e30f; int bp = 0;
    for (int t = tid; t < 256 * KNN; t += 256)
      if (cd[t] < best) { best = cd[t]; bp = t; }
    rv[tid] = best; rp[tid] = bp;
    __syncthreads();
    for (int s = 128; s > 0; s >>= 1) {
      if (tid < s && rv[tid + s] < rv[tid]) { rv[tid] = rv[tid + s]; rp[tid] = rp[tid + s]; }
      __syncthreads();
    }
    if (tid == 0) { idx[(size_t)bn * KNN + sel] = ci[rp[0]]; cd[rp[0]] = 2e30f; }
    __syncthreads();
  }
}

// ---- hg = ht = h --------------------------------------------------------
__global__ void copy2_k(const float* __restrict__ s, float* __restrict__ d1,
                        float* __restrict__ d2, int n) {
  int i = blockIdx.x * 256 + threadIdx.x;
  if (i < n) { float v = s[i]; d1[i] = v; d2[i] = v; }
}

// ---- GCN: o = mean-of-9-neighbors(s); hg += leaky(LN(o)*g+beta) ---------
__global__ __launch_bounds__(256) void gcn_post_k(
    const float* __restrict__ s, const int* __restrict__ idx,
    const float* __restrict__ g, const float* __restrict__ beta,
    float* __restrict__ hg) {
  __shared__ float o[HID];
  __shared__ float red[256];
  int bn = blockIdx.x, b = bn / Nn;
  const float* sb = s + (size_t)b * Nn * HID;
  const int* ix = idx + (size_t)bn * KNN;
  int i0[KNN];
  #pragma unroll
  for (int j = 0; j < KNN; ++j) i0[j] = ix[j];
  for (int c = threadIdx.x; c < HID; c += 256) {
    float acc = 0.f;
    #pragma unroll
    for (int j = 0; j < KNN; ++j) acc += sb[(size_t)i0[j] * HID + c];
    o[c] = acc * (1.f / (float)KNN);          // dinv_n * dinv_m == 1/9 (degree const)
  }
  __syncthreads();
  float p = 0.f, p2 = 0.f;
  for (int c = threadIdx.x; c < HID; c += 256) { p += o[c]; p2 += o[c] * o[c]; }
  float mean = blk_sum256(p, red) * (1.f / HID);
  float ex2  = blk_sum256(p2, red) * (1.f / HID);
  float rstd = rsqrtf(fmaxf(ex2 - mean * mean, 0.f) + EPS);
  for (int c = threadIdx.x; c < HID; c += 256) {
    float y = (o[c] - mean) * rstd * g[c] + beta[c];
    y = (y > 0.f) ? y : 0.2f * y;
    hg[(size_t)bn * HID + c] += y;
  }
}

// ---- post-LN residual: x = LN(x + r)*g + b ------------------------------
__global__ __launch_bounds__(256) void ln_res_k(float* __restrict__ x,
    const float* __restrict__ r, const float* __restrict__ g,
    const float* __restrict__ bta) {
  __shared__ float tt[HID];
  __shared__ float red[256];
  int row = blockIdx.x;
  for (int c = threadIdx.x; c < HID; c += 256)
    tt[c] = x[(size_t)row * HID + c] + r[(size_t)row * HID + c];
  __syncthreads();
  float p = 0.f, p2 = 0.f;
  for (int c = threadIdx.x; c < HID; c += 256) { p += tt[c]; p2 += tt[c] * tt[c]; }
  float mean = blk_sum256(p, red) * (1.f / HID);
  float ex2  = blk_sum256(p2, red) * (1.f / HID);
  float rstd = rsqrtf(fmaxf(ex2 - mean * mean, 0.f) + EPS);
  for (int c = threadIdx.x; c < HID; c += 256)
    x[(size_t)row * HID + c] = (tt[c] - mean) * rstd * g[c] + bta[c];
}

// ---- concat [hg|ht] -----------------------------------------------------
__global__ void concat_k(const float* __restrict__ hg, const float* __restrict__ ht,
                         float* __restrict__ hc) {
  int i = blockIdx.x * 256 + threadIdx.x;    // MT*1024
  int r = i >> 10, c = i & 1023;
  hc[i] = (c < HID) ? hg[(size_t)r * HID + c] : ht[(size_t)r * HID + c - HID];
}

// ---- pooling-attention score: sc = t1 . w2 + b2 -------------------------
__global__ __launch_bounds__(256) void ascore_k(const float* __restrict__ t1,
    const float* __restrict__ w2, const float* __restrict__ b2,
    float* __restrict__ sc) {
  __shared__ float red[256];
  int row = blockIdx.x, t = threadIdx.x;
  float s = blk_sum256(t1[(size_t)row * 256 + t] * w2[t], red);
  if (t == 0) sc[row] = s + b2[0];
}

// ---- softmax over N per batch; writes ws copy and d_out copy ------------
__global__ __launch_bounds__(256) void softmax_k(const float* __restrict__ sc,
    float* __restrict__ aws, float* __restrict__ aout) {
  __shared__ float red[256];
  int b = blockIdx.x;
  const float* si = sc + (size_t)b * Nn;
  float lm = -1e30f;
  for (int n = threadIdx.x; n < Nn; n += 256) lm = fmaxf(lm, si[n]);
  float gm = blk_max256(lm, red);
  float ls = 0.f;
  for (int n = threadIdx.x; n < Nn; n += 256) ls += expf(si[n] - gm);
  float gs = blk_sum256(ls, red);
  float inv = 1.f / gs;
  for (int n = threadIdx.x; n < Nn; n += 256) {
    float v = expf(si[n] - gm) * inv;
    aws[(size_t)b * Nn + n]  = v;
    aout[(size_t)b * Nn + n] = v;
  }
}

// ---- M[b,c] = sum_n hf * a ----------------------------------------------
__global__ __launch_bounds__(256) void pool_k(const float* __restrict__ hf,
    const float* __restrict__ aws, float* __restrict__ Mp,
    float* __restrict__ mout) {
  __shared__ float red[256];
  int o = blockIdx.x, b = o / HID, c = o % HID;
  float p = 0.f;
  for (int n = threadIdx.x; n < Nn; n += 256)
    p += hf[((size_t)b * Nn + n) * HID + c] * aws[(size_t)b * Nn + n];
  float s = blk_sum256(p, red);
  if (threadIdx.x == 0) { Mp[o] = s; mout[o] = s; }
}

// ---- classifier: elu(M@w1^T+b1)@w2^T+b2 --------------------------------
__global__ __launch_bounds__(256) void cls_k(const float* __restrict__ Mp,
    const float* __restrict__ w1, const float* __restrict__ b1,
    const float* __restrict__ w2, const float* __restrict__ b2,
    float* __restrict__ out) {
  __shared__ float red[256];
  int b = blockIdx.x, t = threadIdx.x;
  float acc = 0.f;
  for (int c = 0; c < HID; ++c) acc += Mp[(size_t)b * HID + c] * w1[(size_t)t * HID + c];
  acc += b1[t];
  float e = (acc > 0.f) ? acc : (expf(acc) - 1.f);
  float s = blk_sum256(e * w2[t], red);
  if (t == 0) out[b] = s + b2[0];
}

// =========================================================================
extern "C" void kernel_launch(void* const* d_in, const int* in_sizes, int n_in,
                              void* d_out, int out_size, void* d_ws, size_t ws_size,
                              hipStream_t stream) {
  (void)in_sizes; (void)n_in; (void)out_size; (void)ws_size;
  const float* x       = (const float*)d_in[0];
  const float* w_start = (const float*)d_in[1];
  const float* b_start = (const float*)d_in[2];
  const float* glu_w   = (const float*)d_in[3];
  const float* glu_b   = (const float*)d_in[4];
  const float* gcn_w   = (const float*)d_in[5];
  const float* gcn_b   = (const float*)d_in[6];
  const float* gcn_g   = (const float*)d_in[7];
  const float* gcn_be  = (const float*)d_in[8];
  const float* qkv_w   = (const float*)d_in[9];
  const float* qkv_b   = (const float*)d_in[10];
  const float* out_w   = (const float*)d_in[11];
  const float* out_b   = (const float*)d_in[12];
  const float* ln1_g   = (const float*)d_in[13];
  const float* ln1_b   = (const float*)d_in[14];
  const float* ffn_w1  = (const float*)d_in[15];
  const float* ffn_b1  = (const float*)d_in[16];
  const float* ffn_w2  = (const float*)d_in[17];
  const float* ffn_b2  = (const float*)d_in[18];
  const float* ln2_g   = (const float*)d_in[19];
  const float* ln2_b   = (const float*)d_in[20];
  const float* fus_w   = (const float*)d_in[21];
  const float* fus_b   = (const float*)d_in[22];
  const float* attn_w1 = (const float*)d_in[23];
  const float* attn_b1 = (const float*)d_in[24];
  const float* attn_w2 = (const float*)d_in[25];
  const float* attn_b2 = (const float*)d_in[26];
  const float* cls_w1  = (const float*)d_in[27];
  const float* cls_b1  = (const float*)d_in[28];
  const float* cls_w2  = (const float*)d_in[29];
  const float* cls_b2  = (const float*)d_in[30];
  float* out = (float*)d_out;

  // workspace layout (floats), with lifetime-based aliasing
  float* Wf = (float*)d_ws;
  const size_t M1 = (size_t)1024 * 1024;
  float* h0    = Wf;                // [4096,512]  -> reused as sbuf (gcn s / attn proj / ffn out)
  float* vg    = Wf + 2 * M1;       // [4096,1024] -> reused as big (ffn hid / concat / t1)
  float* hbuf  = Wf + 6 * M1;       // [4096,512]  -> reused as attention output
  float* hg    = Wf + 8 * M1;
  float* ht    = Wf + 10 * M1;
  float* qkvb  = Wf + 12 * M1;      // [4096,1536]
  float* hf    = Wf + 18 * M1;      // [4096,512]
  float* sq    = Wf + 20 * M1;      // [4096]
  float* score = sq + 4096;         // [4096]
  float* aws   = score + 4096;      // [4096]
  float* Mp    = aws + 4096;        // [1024]
  int*   idx   = (int*)(Mp + 1024); // [4096*9]
  float* sbuf = h0;
  float* big  = vg;
  float* aob  = hbuf;

  dim3 blk128(128), blk256(256), blk32(32);

  // fc_start + GLU
  gemm_xwT<0><<<dim3(HID / 64, MT / 64), blk128, 0, stream>>>(x, w_start, b_start, h0, MT, HID, IND);
  gemm_xwT<0><<<dim3((2 * HID) / 64, MT / 64), blk128, 0, stream>>>(h0, glu_w, glu_b, vg, MT, 2 * HID, HID);
  glu_k<<<(MT * HID) / 256, blk256, 0, stream>>>(vg, hbuf);

  // dynamic KNN graph (constant degree 9 -> adjacency is uniform 1/9 gather)
  rowsq_k<<<MT, blk256, 0, stream>>>(hbuf, sq);
  knn_k<<<MT, blk256, 0, stream>>>(hbuf, sq, idx);
  copy2_k<<<(MT * HID) / 256, blk256, 0, stream>>>(hbuf, hg, ht, MT * HID);

  // GCN branch
  for (int i = 0; i < 2; ++i) {
    gemm_xwT<0><<<dim3(HID / 64, MT / 64), blk128, 0, stream>>>(
        hg, gcn_w + (size_t)i * HID * HID, gcn_b + i * HID, sbuf, MT, HID, HID);
    gcn_post_k<<<MT, blk256, 0, stream>>>(sbuf, idx, gcn_g + i * HID, gcn_be + i * HID, hg);
  }

  // Transformer branch (post-LN)
  for (int i = 0; i < 2; ++i) {
    gemm_xwT<0><<<dim3((3 * HID) / 64, MT / 64), blk128, 0, stream>>>(
        ht, qkv_w + (size_t)i * 3 * HID * HID, qkv_b + i * 3 * HID, qkvb, MT, 3 * HID, HID);
    flash_k<<<dim3(Nn / 16, HEADS, Bb), blk32, 0, stream>>>(qkvb, aob);
    gemm_xwT<0><<<dim3(HID / 64, MT / 64), blk128, 0, stream>>>(
        aob, out_w + (size_t)i * HID * HID, out_b + i * HID, sbuf, MT, HID, HID);
    ln_res_k<<<MT, blk256, 0, stream>>>(ht, sbuf, ln1_g + i * HID, ln1_b + i * HID);
    gemm_xwT<1><<<dim3((2 * HID) / 64, MT / 64), blk128, 0, stream>>>(
        ht, ffn_w1 + (size_t)i * 2 * HID * HID, ffn_b1 + i * 2 * HID, big, MT, 2 * HID, HID);
    gemm_xwT<0><<<dim3(HID / 64, MT / 64), blk128, 0, stream>>>(
        big, ffn_w2 + (size_t)i * HID * 2 * HID, ffn_b2 + i * HID, sbuf, MT, HID, 2 * HID);
    ln_res_k<<<MT, blk256, 0, stream>>>(ht, sbuf, ln2_g + i * HID, ln2_b + i * HID);
  }

  // fusion + gated attention pooling + classifier
  concat_k<<<(MT * 1024) / 256, blk256, 0, stream>>>(hg, ht, big);
  gemm_xwT<0><<<dim3(HID / 64, MT / 64), blk128, 0, stream>>>(big, fus_w, fus_b, hf, MT, HID, 2 * HID);
  gemm_xwT<2><<<dim3(256 / 64, MT / 64), blk128, 0, stream>>>(hf, attn_w1, attn_b1, big, MT, 256, HID);
  ascore_k<<<MT, blk256, 0, stream>>>(big, attn_w2, attn_b2, score);
  softmax_k<<<Bb, blk256, 0, stream>>>(score, aws, out + 2);
  pool_k<<<Bb * HID, blk256, 0, stream>>>(hf, aws, Mp, out + 2 + Bb * Nn);
  cls_k<<<Bb, blk256, 0, stream>>>(Mp, cls_w1, cls_b1, cls_w2, cls_b2, out);
}
